// VGG_Cifar10_69965017252751
// MI455X (gfx1250) — 8-bit matrix operand order repaired and checked by reading, compiled, not run on hardware
//
#include <hip/hip_runtime.h>
#include <stdint.h>

typedef int v8i __attribute__((ext_vector_type(8)));

#define BN_EPS 1e-5f

__device__ __forceinline__ int imax2(int a, int b) { return a > b ? a : b; }

__device__ __forceinline__ int8_t quant3(float v) {
  v = fminf(1.0f, fmaxf(-1.0f, v));
  return (int8_t)__float2int_rn(v * 3.0f);
}

__global__ __launch_bounds__(256) void quantize_w_kernel(const float* __restrict__ w,
                                                         int8_t* __restrict__ q, int n) {
  int i = blockIdx.x * 256 + threadIdx.x;
  if (i < n) q[i] = quant3(w[i]);
}

__global__ __launch_bounds__(256) void quantize_conv_w_kernel(const float* __restrict__ w,
                                                              int8_t* __restrict__ q,
                                                              int Cin, int n) {
  int idx = blockIdx.x * 256 + threadIdx.x;
  if (idx >= n) return;
  int ci = idx % Cin;
  int t = idx / Cin;
  int rc = t % 9;
  int o = t / 9;
  q[idx] = quant3(w[(o * Cin + ci) * 9 + rc]);
}

__global__ __launch_bounds__(256) void quantize_fc1_kernel(const float* __restrict__ w,
                                                           int8_t* __restrict__ q) {
  int idx = blockIdx.x * 256 + threadIdx.x;
  int col = idx & 8191;
  int o = idx >> 13;
  int c = col & 511;
  int hw = col >> 9;
  q[idx] = quant3(w[o * 8192 + c * 16 + hw]);
}

__global__ __launch_bounds__(256) void conv1_f32_kernel(const float* __restrict__ x,
                                                        const int8_t* __restrict__ wq,
                                                        float* __restrict__ out) {
  const int H = 32, W = 32;
  int idx = blockIdx.x * 256 + threadIdx.x;
  int co = idx & 127;
  int t = idx >> 7;
  int w = t & 31;
  t >>= 5;
  int h = t & 31;
  int n = t >> 5;
  float acc = 0.f;
#pragma unroll
  for (int ci = 0; ci < 3; ++ci)
#pragma unroll
    for (int r = 0; r < 3; ++r)
#pragma unroll
      for (int c = 0; c < 3; ++c) {
        int hh = h + r - 1, ww = w + c - 1;
        if (hh >= 0 && hh < H && ww >= 0 && ww < W)
          acc += x[((n * 3 + ci) * H + hh) * W + ww] *
                 (float)wq[((co * 3 + ci) * 3 + r) * 3 + c];
      }
  out[idx] = acc * (1.0f / 3.0f);
}

__global__ __launch_bounds__(256) void conv3x3_wmma_iu8_kernel(
    const int8_t* __restrict__ act, const int8_t* __restrict__ wq,
    int* __restrict__ out, int Cin, int Cout, int H, int W) {
  const int K = Cin * 9;
  const int HW = H * W;
  const int pixel0 = blockIdx.x * 32;
  const int cout0 = blockIdx.y * 128;
  const int tid = threadIdx.x;
  const int lane = tid & 31;
  const int wv = tid >> 5;
  const int msub = (wv & 3) * 32;
  const int psub = (wv >> 2) * 16;
  const int hf = lane >> 4;
  const int l16 = lane & 15;

  const int pix = pixel0 + psub + l16;
  const int n = pix / HW;
  const int prem = pix - n * HW;
  const int ph = prem / W;
  const int pw = prem - ph * W;

  const int8_t* b0 = wq + (size_t)(cout0 + msub + l16) * K;
  const int8_t* b1 = b0 + (size_t)16 * K;

  v8i acc0 = {0, 0, 0, 0, 0, 0, 0, 0};
  v8i acc1 = {0, 0, 0, 0, 0, 0, 0, 0};

#pragma unroll
  for (int rc = 0; rc < 9; ++rc) {
    const int dh = rc / 3 - 1;
    const int dw = rc % 3 - 1;
    const int hh = ph + dh, ww = pw + dw;
    const bool valid = (hh >= 0) && (hh < H) && (ww >= 0) && (ww < W);
    const int8_t* arow =
        valid ? act + ((size_t)(n * H + hh) * W + ww) * Cin : act;
    for (int ci = 0; ci < Cin; ci += 64) {
      const int kb = rc * Cin + ci;
      v8i af, bf0, bf1;
#pragma unroll
      for (int i = 0; i < 8; ++i) {
        const int ka = ((i >> 1) << 4) + ((i & 1) << 2) + (hf << 3);
        int a = *(const int*)(arow + ci + ka);
        af[i] = valid ? a : 0;
        const int kbb = ((i >> 1) << 4) + ((i & 1) << 2) + (hf << 3);
        bf0[i] = *(const int*)(b0 + kb + kbb);
        bf1[i] = *(const int*)(b1 + kb + kbb);
      }
      acc0 = __builtin_amdgcn_wmma_i32_16x16x64_iu8(true, af, true, bf0, acc0, false, false);
      acc1 = __builtin_amdgcn_wmma_i32_16x16x64_iu8(true, af, true, bf1, acc1, false, false);
    }
  }

#pragma unroll
  for (int r = 0; r < 8; ++r) {
    int* orow = out + (size_t)(pixel0 + psub + r + hf * 8) * Cout;
    orow[cout0 + msub + l16] = acc0[r];
    orow[cout0 + msub + 16 + l16] = acc1[r];
  }
}

__global__ __launch_bounds__(256) void gemm_iu8_kernel(const int8_t* __restrict__ A,
                                                       const int8_t* __restrict__ Bw,
                                                       int* __restrict__ D,
                                                       int M, int N, int K) {
  const int tid = threadIdx.x;
  const int lane = tid & 31;
  const int wv = tid >> 5;
  const int hf = lane >> 4;
  const int l16 = lane & 15;
  const int m0 = blockIdx.y * 32 + (wv & 1) * 16;
  const int n0 = blockIdx.x * 128 + (wv >> 1) * 32;
  const int nrow0 = n0 + l16;
  const int nrow1 = n0 + 16 + l16;
  const bool nv0 = nrow0 < N;
  const bool nv1 = nrow1 < N;
  const int8_t* ap = A + (size_t)(m0 + l16) * K;
  const int8_t* bp0 = Bw + (nv0 ? (size_t)nrow0 * K : 0);
  const int8_t* bp1 = Bw + (nv1 ? (size_t)nrow1 * K : 0);
  v8i acc0 = {0, 0, 0, 0, 0, 0, 0, 0};
  v8i acc1 = {0, 0, 0, 0, 0, 0, 0, 0};
  for (int kb = 0; kb < K; kb += 64) {
    v8i af, bf0, bf1;
#pragma unroll
    for (int i = 0; i < 8; ++i) {
      const int ka = ((i >> 1) << 4) + ((i & 1) << 2) + (hf << 3);
      af[i] = *(const int*)(ap + kb + ka);
      const int kbb = ((i >> 1) << 4) + ((i & 1) << 2) + (hf << 3);
      bf0[i] = nv0 ? *(const int*)(bp0 + kb + kbb) : 0;
      bf1[i] = nv1 ? *(const int*)(bp1 + kb + kbb) : 0;
    }
    if (kb + 64 < K) {
      __builtin_prefetch(ap + kb + 64, 0, 1);
      __builtin_prefetch(bp0 + kb + 64, 0, 1);
      __builtin_prefetch(bp1 + kb + 64, 0, 1);
    }
    acc0 = __builtin_amdgcn_wmma_i32_16x16x64_iu8(true, af, true, bf0, acc0, false, false);
    acc1 = __builtin_amdgcn_wmma_i32_16x16x64_iu8(true, af, true, bf1, acc1, false, false);
  }
#pragma unroll
  for (int r = 0; r < 8; ++r) {
    int m = m0 + r + hf * 8;
    if (m < M) {
      if (nv0) D[(size_t)m * N + nrow0] = acc0[r];
      if (nv1) D[(size_t)m * N + nrow1] = acc1[r];
    }
  }
}

__global__ __launch_bounds__(256) void maxpool2_i32_kernel(const int* __restrict__ in,
                                                           int* __restrict__ out,
                                                           int C, int H, int W) {
  const int Ho = H >> 1, Wo = W >> 1;
  int idx = blockIdx.x * 256 + threadIdx.x;
  int c = idx % C;
  int t = idx / C;
  int wo = t % Wo;
  t /= Wo;
  int ho = t % Ho;
  int n = t / Ho;
  const int* p = in + ((size_t)(n * H + 2 * ho) * W + 2 * wo) * C + c;
  const size_t rs = (size_t)W * C;
  out[idx] = imax2(imax2(p[0], p[C]), imax2(p[rs], p[rs + C]));
}

__global__ __launch_bounds__(256) void bn_stats_kernel(const void* __restrict__ src,
                                                       int isInt, float scale, int C,
                                                       long count,
                                                       float* __restrict__ mean,
                                                       float* __restrict__ rstd) {
  const int c = blockIdx.x;
  const int tid = threadIdx.x;
  float s = 0.f, sq = 0.f;
  for (long i = tid; i < count; i += 256) {
    long a = i * C + c;
    float v = isInt ? (float)((const int*)src)[a] * scale : ((const float*)src)[a];
    s += v;
    sq += v * v;
  }
  __shared__ float ls[256], lq[256];
  ls[tid] = s;
  lq[tid] = sq;
  __syncthreads();
  for (int st = 128; st > 0; st >>= 1) {
    if (tid < st) { ls[tid] += ls[tid + st]; lq[tid] += lq[tid + st]; }
    __syncthreads();
  }
  if (tid == 0) {
    float m = ls[0] / (float)count;
    float var = lq[0] / (float)count - m * m;
    mean[c] = m;
    rstd[c] = rsqrtf(var + BN_EPS);
  }
}

__global__ __launch_bounds__(256) void bn_htanh_sign_kernel(
    const void* __restrict__ src, int isInt, float scale, int Cmask,
    const float* __restrict__ mean, const float* __restrict__ rstd,
    const float* __restrict__ g, const float* __restrict__ b,
    int8_t* __restrict__ out, long total) {
  long idx = (long)blockIdx.x * 256 + threadIdx.x;
  if (idx >= total) return;
  int c = (int)(idx & Cmask);
  float v = isInt ? (float)((const int*)src)[idx] * scale : ((const float*)src)[idx];
  float y = (v - mean[c]) * rstd[c] * g[c] + b[c];
  out[idx] = (int8_t)((y > 0.f) ? 1 : ((y < 0.f) ? -1 : 0));
}

__global__ __launch_bounds__(256) void logsoftmax_kernel(const int* __restrict__ fc3,
                                                         const float* __restrict__ mean,
                                                         const float* __restrict__ rstd,
                                                         float* __restrict__ out) {
  int s = blockIdx.x * 256 + threadIdx.x;
  if (s >= 512) return;
  float y[10];
  float mx = -1e30f;
#pragma unroll
  for (int j = 0; j < 10; ++j) {
    float v = (float)fc3[s * 10 + j] * (1.0f / 3.0f);
    y[j] = (v - mean[j]) * rstd[j];
    mx = fmaxf(mx, y[j]);
  }
  float se = 0.f;
#pragma unroll
  for (int j = 0; j < 10; ++j) se += expf(y[j] - mx);
  float l = mx + logf(se);
#pragma unroll
  for (int j = 0; j < 10; ++j) out[s * 10 + j] = y[j] - l;
}

extern "C" void kernel_launch(void* const* d_in, const int* in_sizes, int n_in,
                              void* d_out, int out_size, void* d_ws, size_t ws_size,
                              hipStream_t stream) {
  (void)n_in; (void)out_size; (void)ws_size;
  const float* x = (const float*)d_in[0];
  const float* g[9];
  const float* bb[9];
  for (int i = 0; i < 8; ++i) {
    g[i + 1] = (const float*)d_in[10 + 2 * i];
    bb[i + 1] = (const float*)d_in[11 + 2 * i];
  }

  char* ws = (char*)d_ws;
  size_t off = 0;
  auto carve = [&](size_t bytes) -> void* {
    void* p = ws + off;
    off += (bytes + 255) & ~(size_t)255;
    return p;
  };
  int8_t* qw[10];
  for (int i = 1; i <= 9; ++i) qw[i] = (int8_t*)carve((size_t)in_sizes[i]);
  const size_t ACT = (size_t)512 * 128 * 32 * 32;
  int8_t* actA = (int8_t*)carve(ACT);
  int8_t* actB = (int8_t*)carve(ACT);
  int* accum = (int*)carve(ACT * 4);
  int* pooled = (int*)carve((size_t)512 * 128 * 16 * 16 * 4);
  int* fcbuf = (int*)carve((size_t)512 * 1024 * 4);
  float* mean = (float*)carve(1024 * 4);
  float* rstd = (float*)carve(1024 * 4);

  quantize_w_kernel<<<(in_sizes[1] + 255) / 256, 256, 0, stream>>>((const float*)d_in[1], qw[1], in_sizes[1]);
  const int convCin[7] = {0, 0, 128, 128, 256, 256, 512};
  for (int i = 2; i <= 6; ++i)
    quantize_conv_w_kernel<<<(in_sizes[i] + 255) / 256, 256, 0, stream>>>(
        (const float*)d_in[i], qw[i], convCin[i], in_sizes[i]);
  quantize_fc1_kernel<<<(1024 * 8192) / 256, 256, 0, stream>>>((const float*)d_in[7], qw[7]);
  quantize_w_kernel<<<(in_sizes[8] + 255) / 256, 256, 0, stream>>>((const float*)d_in[8], qw[8], in_sizes[8]);
  quantize_w_kernel<<<(in_sizes[9] + 255) / 256, 256, 0, stream>>>((const float*)d_in[9], qw[9], in_sizes[9]);

  conv1_f32_kernel<<<(512 * 128 * 32 * 32) / 256, 256, 0, stream>>>(x, qw[1], (float*)accum);
  bn_stats_kernel<<<128, 256, 0, stream>>>(accum, 0, 1.f, 128, (long)512 * 1024, mean, rstd);
  bn_htanh_sign_kernel<<<(512 * 128 * 1024) / 256, 256, 0, stream>>>(
      accum, 0, 1.f, 127, mean, rstd, g[1], bb[1], actA, (long)512 * 128 * 1024);

  conv3x3_wmma_iu8_kernel<<<dim3(512 * 32 * 32 / 32, 1), 256, 0, stream>>>(
      actA, qw[2], accum, 128, 128, 32, 32);
  maxpool2_i32_kernel<<<(512 * 128 * 16 * 16) / 256, 256, 0, stream>>>(accum, pooled, 128, 32, 32);
  bn_stats_kernel<<<128, 256, 0, stream>>>(pooled, 1, 1.f / 3.f, 128, (long)512 * 256, mean, rstd);
  bn_htanh_sign_kernel<<<(512 * 128 * 256) / 256, 256, 0, stream>>>(
      pooled, 1, 1.f / 3.f, 127, mean, rstd, g[2], bb[2], actB, (long)512 * 128 * 256);

  conv3x3_wmma_iu8_kernel<<<dim3(512 * 16 * 16 / 32, 2), 256, 0, stream>>>(
      actB, qw[3], accum, 128, 256, 16, 16);
  bn_stats_kernel<<<256, 256, 0, stream>>>(accum, 1, 1.f / 3.f, 256, (long)512 * 256, mean, rstd);
  bn_htanh_sign_kernel<<<(512 * 256 * 256) / 256, 256, 0, stream>>>(
      accum, 1, 1.f / 3.f, 255, mean, rstd, g[3], bb[3], actA, (long)512 * 256 * 256);

  conv3x3_wmma_iu8_kernel<<<dim3(512 * 16 * 16 / 32, 2), 256, 0, stream>>>(
      actA, qw[4], accum, 256, 256, 16, 16);
  maxpool2_i32_kernel<<<(512 * 256 * 8 * 8) / 256, 256, 0, stream>>>(accum, pooled, 256, 16, 16);
  bn_stats_kernel<<<256, 256, 0, stream>>>(pooled, 1, 1.f / 3.f, 256, (long)512 * 64, mean, rstd);
  bn_htanh_sign_kernel<<<(512 * 256 * 64) / 256, 256, 0, stream>>>(
      pooled, 1, 1.f / 3.f, 255, mean, rstd, g[4], bb[4], actB, (long)512 * 256 * 64);

  conv3x3_wmma_iu8_kernel<<<dim3(512 * 8 * 8 / 32, 4), 256, 0, stream>>>(
      actB, qw[5], accum, 256, 512, 8, 8);
  bn_stats_kernel<<<512, 256, 0, stream>>>(accum, 1, 1.f / 3.f, 512, (long)512 * 64, mean, rstd);
  bn_htanh_sign_kernel<<<(512 * 512 * 64) / 256, 256, 0, stream>>>(
      accum, 1, 1.f / 3.f, 511, mean, rstd, g[5], bb[5], actA, (long)512 * 512 * 64);

  conv3x3_wmma_iu8_kernel<<<dim3(512 * 8 * 8 / 32, 4), 256, 0, stream>>>(
      actA, qw[6], accum, 512, 512, 8, 8);
  maxpool2_i32_kernel<<<(512 * 512 * 4 * 4) / 256, 256, 0, stream>>>(accum, pooled, 512, 8, 8);
  bn_stats_kernel<<<512, 256, 0, stream>>>(pooled, 1, 1.f / 3.f, 512, (long)512 * 16, mean, rstd);
  bn_htanh_sign_kernel<<<(512 * 512 * 16) / 256, 256, 0, stream>>>(
      pooled, 1, 1.f / 3.f, 511, mean, rstd, g[6], bb[6], actB, (long)512 * 512 * 16);

  gemm_iu8_kernel<<<dim3(1024 / 128, 512 / 32), 256, 0, stream>>>(actB, qw[7], fcbuf, 512, 1024, 8192);
  bn_stats_kernel<<<1024, 256, 0, stream>>>(fcbuf, 1, 1.f / 3.f, 1024, (long)512, mean, rstd);
  bn_htanh_sign_kernel<<<(512 * 1024) / 256, 256, 0, stream>>>(
      fcbuf, 1, 1.f / 3.f, 1023, mean, rstd, g[7], bb[7], actA, (long)512 * 1024);

  gemm_iu8_kernel<<<dim3(1024 / 128, 512 / 32), 256, 0, stream>>>(actA, qw[8], fcbuf, 512, 1024, 1024);
  bn_stats_kernel<<<1024, 256, 0, stream>>>(fcbuf, 1, 1.f / 3.f, 1024, (long)512, mean, rstd);
  bn_htanh_sign_kernel<<<(512 * 1024) / 256, 256, 0, stream>>>(
      fcbuf, 1, 1.f / 3.f, 1023, mean, rstd, g[8], bb[8], actB, (long)512 * 1024);

  gemm_iu8_kernel<<<dim3(1, 512 / 32), 256, 0, stream>>>(actB, qw[9], fcbuf, 512, 10, 1024);
  bn_stats_kernel<<<10, 256, 0, stream>>>(fcbuf, 1, 1.f / 3.f, 10, (long)512, mean, rstd);
  logsoftmax_kernel<<<2, 256, 0, stream>>>(fcbuf, mean, rstd, (float*)d_out);
}
